// FeedForwardModule_67439576482273
// MI455X (gfx1250) — 8-bit matrix operand order repaired and checked by reading, compiled, not run on hardware
//
#include <hip/hip_runtime.h>

typedef __attribute__((ext_vector_type(8))) int v8i;

#define FF_EPS 1e-5f

#define M_TOK 16384
#define D_DIM 1024
#define F_DIM 4096

__device__ __forceinline__ v8i load_a_frag(const signed char* p) {
  v8i a;
  int2 c0 = *reinterpret_cast<const int2*>(p);
  int2 c1 = *reinterpret_cast<const int2*>(p + 8);
  int2 c2 = *reinterpret_cast<const int2*>(p + 32);
  int2 c3 = *reinterpret_cast<const int2*>(p + 40);
  a[0] = c0.x; a[1] = c0.y; a[2] = c1.x; a[3] = c1.y;
  a[4] = c2.x; a[5] = c2.y; a[6] = c3.x; a[7] = c3.y;
  return a;
}

__device__ __forceinline__ v8i load_b_frag(const signed char* p) {
  v8i b;
  int4 c0 = *reinterpret_cast<const int4*>(p);
  int4 c1 = *reinterpret_cast<const int4*>(p + 32);
  b[0] = c0.x; b[1] = c0.y; b[2] = c0.z; b[3] = c0.w;
  b[4] = c1.x; b[5] = c1.y; b[6] = c1.z; b[7] = c1.w;
  return b;
}

__global__ void k_init(unsigned* __restrict__ hmax, float* __restrict__ w1m,
                       float* __restrict__ w2m) {
  int i = blockIdx.x * blockDim.x + threadIdx.x;
  if (i == 0) { *w1m = 0.f; *w2m = 0.f; }
  for (; i < M_TOK; i += gridDim.x * blockDim.x) hmax[i] = 0u;
}

__global__ void k_absmax(const float* __restrict__ w, size_t n,
                         float* __restrict__ out) {
  __shared__ float red[256];
  float m = 0.f;
  for (size_t i = (size_t)blockIdx.x * blockDim.x + threadIdx.x; i < n;
       i += (size_t)gridDim.x * blockDim.x)
    m = fmaxf(m, fabsf(w[i]));
  red[threadIdx.x] = m;
  __syncthreads();
  for (int off = 128; off > 0; off >>= 1) {
    if ((int)threadIdx.x < off)
      red[threadIdx.x] = fmaxf(red[threadIdx.x], red[threadIdx.x + off]);
    __syncthreads();
  }
  if (threadIdx.x == 0)
    atomicMax(reinterpret_cast<unsigned*>(out), __float_as_uint(red[0]));
}

__global__ void k_quant_w(const float* __restrict__ w,
                          signed char* __restrict__ qw,
                          const float* __restrict__ amax,
                          const int* __restrict__ bits, size_t n) {
  const float qb = (float)((1 << (*bits - 1)) - 1);
  const float scale = qb / fmaxf(*amax, FF_EPS);
  for (size_t i = (size_t)blockIdx.x * blockDim.x + threadIdx.x; i < n;
       i += (size_t)gridDim.x * blockDim.x) {
    float t = fminf(fmaxf(w[i] * scale, -qb), qb);
    qw[i] = (signed char)rintf(t);
  }
}

__global__ void k_ln_quant(const float* __restrict__ x,
                           const float* __restrict__ gamma,
                           const float* __restrict__ beta,
                           signed char* __restrict__ q,
                           float* __restrict__ a_rs) {
  __shared__ float red[256];
  const int tid = threadIdx.x;
  const int row = blockIdx.x;
  const float* xr = x + (size_t)row * D_DIM;

  float v[4];
  float s = 0.f;
#pragma unroll
  for (int j = 0; j < 4; ++j) { v[j] = xr[tid + 256 * j]; s += v[j]; }
  red[tid] = s;
  __syncthreads();
  for (int off = 128; off > 0; off >>= 1) {
    if (tid < off) red[tid] += red[tid + off];
    __syncthreads();
  }
  const float mu = red[0] * (1.f / (float)D_DIM);
  __syncthreads();

  float s2 = 0.f;
#pragma unroll
  for (int j = 0; j < 4; ++j) { float d = v[j] - mu; s2 += d * d; }
  red[tid] = s2;
  __syncthreads();
  for (int off = 128; off > 0; off >>= 1) {
    if (tid < off) red[tid] += red[tid + off];
    __syncthreads();
  }
  const float rstd = rsqrtf(red[0] * (1.f / (float)D_DIM) + FF_EPS);
  __syncthreads();

  float nv[4];
  float am = 0.f;
#pragma unroll
  for (int j = 0; j < 4; ++j) {
    const int idx = tid + 256 * j;
    nv[j] = (v[j] - mu) * rstd * gamma[idx] + beta[idx];
    am = fmaxf(am, fabsf(nv[j]));
  }
  red[tid] = am;
  __syncthreads();
  for (int off = 128; off > 0; off >>= 1) {
    if (tid < off) red[tid] = fmaxf(red[tid], red[tid + off]);
    __syncthreads();
  }
  const float amax = fmaxf(red[0], FF_EPS);
  const float scale = 127.f / amax;
#pragma unroll
  for (int j = 0; j < 4; ++j) {
    float t = fminf(fmaxf(nv[j] * scale, -127.f), 127.f);
    q[(size_t)row * D_DIM + tid + 256 * j] = (signed char)rintf(t);
  }
  if (tid == 0) a_rs[row] = amax * (1.f / 127.f);
}

__global__ __launch_bounds__(256) void k_gemm1(
    const signed char* __restrict__ qa, const signed char* __restrict__ qw,
    const float* __restrict__ a_rs, const float* __restrict__ wmax,
    const int* __restrict__ bits, const float* __restrict__ b1,
    _Float16* __restrict__ h, unsigned* __restrict__ hmax) {
  constexpr int K = D_DIM, N = F_DIM;
  const int lane = threadIdx.x & 31;
  const int wave = threadIdx.x >> 5;
  const int half = lane >> 4;
  const int l15 = lane & 15;
  const int m0 = blockIdx.y * 64 + (wave & 1) * 32;
  const int n0 = blockIdx.x * 256 + (wave >> 1) * 64;

  const signed char* pa0 = qa + (size_t)(m0 + l15) * K + half * 16;
  const signed char* pa1 = pa0 + (size_t)16 * K;
  const signed char* pb = qw + (size_t)(n0 + l15) * K + half * 16;

  v8i acc[8];
#pragma unroll
  for (int j = 0; j < 8; ++j)
#pragma unroll
    for (int i = 0; i < 8; ++i) acc[j][i] = 0;

  for (int k0 = 0; k0 < K; k0 += 64) {
    __builtin_prefetch(pa0 + k0 + 64, 0, 1);
    v8i a0 = load_a_frag(pa0 + k0);
    v8i a1 = load_a_frag(pa1 + k0);
#pragma unroll
    for (int j = 0; j < 4; ++j) {
      v8i b = load_b_frag(pb + (size_t)(j * 16) * K + k0);
      acc[j] = __builtin_amdgcn_wmma_i32_16x16x64_iu8(true, a0, true, b,
                                                      acc[j], false, false);
      acc[4 + j] = __builtin_amdgcn_wmma_i32_16x16x64_iu8(true, a1, true, b,
                                                          acc[4 + j], false,
                                                          false);
    }
  }

  const float qbw = (float)((1 << (*bits - 1)) - 1);
  const float rw = fmaxf(*wmax, FF_EPS) / qbw;
  float ars[16], rmax[16];
#pragma unroll
  for (int mi = 0; mi < 2; ++mi)
#pragma unroll
    for (int i = 0; i < 8; ++i) {
      ars[mi * 8 + i] = a_rs[m0 + mi * 16 + i + half * 8] * rw;
      rmax[mi * 8 + i] = 0.f;
    }
#pragma unroll
  for (int j = 0; j < 4; ++j) {
    const int col = n0 + j * 16 + l15;
    const float bias = b1[col];
#pragma unroll
    for (int mi = 0; mi < 2; ++mi)
#pragma unroll
      for (int i = 0; i < 8; ++i) {
        const int row = m0 + mi * 16 + i + half * 8;
        float y = (float)acc[mi * 4 + j][i] * ars[mi * 8 + i] + bias;
        float sw = y / (1.f + __expf(-y));
        h[(size_t)row * N + col] = (_Float16)sw;
        rmax[mi * 8 + i] = fmaxf(rmax[mi * 8 + i], fabsf(sw));
      }
  }
#pragma unroll
  for (int t = 0; t < 16; ++t) {
    float r = rmax[t];
    r = fmaxf(r, __shfl_xor(r, 1));
    r = fmaxf(r, __shfl_xor(r, 2));
    r = fmaxf(r, __shfl_xor(r, 4));
    r = fmaxf(r, __shfl_xor(r, 8));
    if (l15 == 0) {
      const int row = m0 + (t >> 3) * 16 + (t & 7) + half * 8;
      atomicMax(&hmax[row], __float_as_uint(r));
    }
  }
}

__global__ void k_quant_h(const _Float16* __restrict__ h,
                          const unsigned* __restrict__ hmax,
                          signed char* __restrict__ qh) {
  const size_t quads = (size_t)M_TOK * (F_DIM / 4);
  for (size_t i = (size_t)blockIdx.x * blockDim.x + threadIdx.x; i < quads;
       i += (size_t)gridDim.x * blockDim.x) {
    const size_t row = i >> 10;
    const float scale = 127.f / fmaxf(__uint_as_float(hmax[row]), FF_EPS);
    const _Float16* hp = h + i * 4;
    signed char o[4];
#pragma unroll
    for (int k = 0; k < 4; ++k) {
      float t = fminf(fmaxf((float)hp[k] * scale, -127.f), 127.f);
      o[k] = (signed char)rintf(t);
    }
    *reinterpret_cast<char4*>(qh + i * 4) = make_char4(o[0], o[1], o[2], o[3]);
  }
}

__global__ __launch_bounds__(256) void k_gemm2(
    const signed char* __restrict__ qh, const signed char* __restrict__ qw,
    const unsigned* __restrict__ hmax, const float* __restrict__ wmax,
    const int* __restrict__ bits, const float* __restrict__ b2,
    const float* __restrict__ x, const float* __restrict__ mask,
    float* __restrict__ out) {
  constexpr int K = F_DIM, N = D_DIM;
  const int lane = threadIdx.x & 31;
  const int wave = threadIdx.x >> 5;
  const int half = lane >> 4;
  const int l15 = lane & 15;
  const int m0 = blockIdx.y * 64 + (wave & 1) * 32;
  const int n0 = blockIdx.x * 256 + (wave >> 1) * 64;

  const signed char* pa0 = qh + (size_t)(m0 + l15) * K + half * 16;
  const signed char* pa1 = pa0 + (size_t)16 * K;
  const signed char* pb = qw + (size_t)(n0 + l15) * K + half * 16;

  v8i acc[8];
#pragma unroll
  for (int j = 0; j < 8; ++j)
#pragma unroll
    for (int i = 0; i < 8; ++i) acc[j][i] = 0;

  for (int k0 = 0; k0 < K; k0 += 64) {
    __builtin_prefetch(pa0 + k0 + 64, 0, 1);
    v8i a0 = load_a_frag(pa0 + k0);
    v8i a1 = load_a_frag(pa1 + k0);
#pragma unroll
    for (int j = 0; j < 4; ++j) {
      v8i b = load_b_frag(pb + (size_t)(j * 16) * K + k0);
      acc[j] = __builtin_amdgcn_wmma_i32_16x16x64_iu8(true, a0, true, b,
                                                      acc[j], false, false);
      acc[4 + j] = __builtin_amdgcn_wmma_i32_16x16x64_iu8(true, a1, true, b,
                                                          acc[4 + j], false,
                                                          false);
    }
  }

  const float qbw = (float)((1 << (*bits - 1)) - 1);
  const float rw = fmaxf(*wmax, FF_EPS) / qbw;
  float hrs[16], msk[16];
#pragma unroll
  for (int mi = 0; mi < 2; ++mi)
#pragma unroll
    for (int i = 0; i < 8; ++i) {
      const int row = m0 + mi * 16 + i + half * 8;
      hrs[mi * 8 + i] =
          fmaxf(__uint_as_float(hmax[row]), FF_EPS) * (1.f / 127.f) * rw;
      msk[mi * 8 + i] = mask[row];
    }
#pragma unroll
  for (int j = 0; j < 4; ++j) {
    const int col = n0 + j * 16 + l15;
    const float bias = b2[col];
#pragma unroll
    for (int mi = 0; mi < 2; ++mi)
#pragma unroll
      for (int i = 0; i < 8; ++i) {
        const int row = m0 + mi * 16 + i + half * 8;
        float y = (float)acc[mi * 4 + j][i] * hrs[mi * 8 + i] + bias;
        y *= msk[mi * 8 + i];
        const size_t idx = (size_t)row * N + col;
        out[idx] = x[idx] + 0.5f * y;
      }
  }
}

extern "C" void kernel_launch(void* const* d_in, const int* in_sizes, int n_in,
                              void* d_out, int out_size, void* d_ws,
                              size_t ws_size, hipStream_t stream) {
  const float* x     = (const float*)d_in[0];
  const float* mask  = (const float*)d_in[1];
  const float* gamma = (const float*)d_in[2];
  const float* beta  = (const float*)d_in[3];
  const float* w1    = (const float*)d_in[4];
  const float* b1    = (const float*)d_in[5];
  const float* w2    = (const float*)d_in[6];
  const float* b2    = (const float*)d_in[7];
  const int*   bits  = (const int*)d_in[8];

  char* ws = (char*)d_ws;
  float*    w1max = (float*)(ws + 0);
  float*    w2max = (float*)(ws + 4);
  float*    a_rs  = (float*)(ws + 256);
  unsigned* hmax  = (unsigned*)(ws + 256 + 4 * M_TOK);
  signed char* qw1  = (signed char*)(ws + 256 + 8 * M_TOK + 256);
  signed char* qw2  = qw1 + (size_t)F_DIM * D_DIM;
  signed char* qact = qw2 + (size_t)D_DIM * F_DIM;
  _Float16*    h    = (_Float16*)(qact + (size_t)M_TOK * D_DIM);
  signed char* qh   = (signed char*)((char*)h + (size_t)M_TOK * F_DIM * 2);

  k_init<<<64, 256, 0, stream>>>(hmax, w1max, w2max);
  k_absmax<<<512, 256, 0, stream>>>(w1, (size_t)F_DIM * D_DIM, w1max);
  k_absmax<<<512, 256, 0, stream>>>(w2, (size_t)D_DIM * F_DIM, w2max);
  k_quant_w<<<1024, 256, 0, stream>>>(w1, qw1, w1max, bits,
                                      (size_t)F_DIM * D_DIM);
  k_quant_w<<<1024, 256, 0, stream>>>(w2, qw2, w2max, bits,
                                      (size_t)D_DIM * F_DIM);
  k_ln_quant<<<M_TOK, 256, 0, stream>>>(x, gamma, beta, qact, a_rs);

  dim3 g1(F_DIM / 256, M_TOK / 64);
  k_gemm1<<<g1, 256, 0, stream>>>(qact, qw1, a_rs, w1max, bits, b1, h, hmax);

  k_quant_h<<<4096, 256, 0, stream>>>(h, hmax, qh);

  dim3 g2(D_DIM / 256, M_TOK / 64);
  k_gemm2<<<g2, 256, 0, stream>>>(qh, qw2, hmax, w2max, bits, b2, x, mask,
                                  (float*)d_out);
}
